// HeadAttention_46162308498202
// MI455X (gfx1250) — matrix operand order repaired and checked by reading, not compiled, not run on hardware
//
#include <hip/hip_runtime.h>
#include <hip/hip_bf16.h>

typedef _Float16 v16h __attribute__((ext_vector_type(16)));
typedef _Float16 v8h  __attribute__((ext_vector_type(8)));
typedef float    v8f  __attribute__((ext_vector_type(8)));
typedef int      v4i  __attribute__((ext_vector_type(4)));

#define EMB   1024
#define HEAD  64
#define SEQ   4096
#define NB    4

#if __has_builtin(__builtin_amdgcn_global_load_async_to_lds_b128)
#define HAVE_ASYNC_LDS 1
#else
#define HAVE_ASYNC_LDS 0
#endif

// A-fragment (16-bit, 16x32): two contiguous 16B chunks (16 k) -> one v16h
__device__ __forceinline__ v16h load_afrag(const _Float16* p) {
    union { v16h v; v8h h[2]; } u;
    u.h[0] = *(const v8h*)p;
    u.h[1] = *(const v8h*)(p + 8);
    return u.v;
}

// async 16-byte global -> LDS copy (falls back to through-VGPR copy)
__device__ __forceinline__ void async_copy16(const _Float16* g, _Float16* l) {
#if HAVE_ASYNC_LDS
    __builtin_amdgcn_global_load_async_to_lds_b128(
        (__attribute__((address_space(1))) v4i*)g,
        (__attribute__((address_space(3))) v4i*)l,
        0, 0);
#else
    *(uint4*)l = *(const uint4*)g;
#endif
}

__device__ __forceinline__ void async_join() {
#if HAVE_ASYNC_LDS
#if __has_builtin(__builtin_amdgcn_s_wait_asynccnt)
    __builtin_amdgcn_s_wait_asynccnt(0);
#else
    asm volatile("s_wait_asynccnt 0x0" ::: "memory");
#endif
#endif
}

// ---------------------------------------------------------------------------
// Kernel 1: fused QKV projection.  x[16384,1024] f32 x W[1024,192] -> f16.
// Q,K stored [token][64]; V stored transposed [b][64][S] for kernel 2 B-frags.
// ---------------------------------------------------------------------------
__global__ __launch_bounds__(256) void qkv_project(
    const float* __restrict__ x,
    const float* __restrict__ Wq, const float* __restrict__ bq,
    const float* __restrict__ Wk, const float* __restrict__ bk,
    const float* __restrict__ Wv, const float* __restrict__ bv,
    _Float16* __restrict__ qb, _Float16* __restrict__ kb,
    _Float16* __restrict__ vt)
{
    __shared__ _Float16 lds_x[128 * 64];   // 16 KB: x chunk [tok][k]
    __shared__ _Float16 lds_w[192 * 64];   // 24 KB: W chunk transposed [n][k]

    const int tid  = threadIdx.x;
    const int wv   = tid >> 5;
    const int lane = tid & 31;
    const int tok0 = blockIdx.x * 128;

    const int mrow   = lane & 15;
    const int ka_off = (lane < 16) ? 0 : 16;   // A-fragment K base
    const int kb_off = (lane < 16) ? 0 : 16;   // B-fragment K base

    v8f zero = {0.f, 0.f, 0.f, 0.f, 0.f, 0.f, 0.f, 0.f};
    v8f acc[12];
    for (int i = 0; i < 12; i++) acc[i] = zero;

    for (int kc = 0; kc < EMB; kc += 64) {
        __syncthreads();
        // stage x chunk (f32 -> f16)
        for (int i = tid; i < 128 * 64; i += 256) {
            int t = i >> 6, k = i & 63;
            lds_x[i] = (_Float16)x[(size_t)(tok0 + t) * EMB + kc + k];
        }
        // stage W chunk transposed: lds_w[n*64+k] = W[kc+k][n]
        for (int i = tid; i < 192 * 64; i += 256) {
            int n = i >> 6, k = i & 63;
            float w;
            if (n < 64)       w = Wq[(size_t)(kc + k) * HEAD + n];
            else if (n < 128) w = Wk[(size_t)(kc + k) * HEAD + (n - 64)];
            else              w = Wv[(size_t)(kc + k) * HEAD + (n - 128)];
            lds_w[i] = (_Float16)w;
        }
        __syncthreads();

        for (int ks = 0; ks < 2; ks++) {
            // A fragment: 16 rows x 32 k
            v16h afrag = load_afrag(&lds_x[(wv * 16 + mrow) * 64 + ks * 32 + ka_off]);
            v16h bf[12];
            for (int nt = 0; nt < 12; nt++)
                bf[nt] = *(const v16h*)&lds_w[(nt * 16 + mrow) * 64 + ks * 32 + kb_off];
            for (int nt = 0; nt < 12; nt++)
                acc[nt] = __builtin_amdgcn_wmma_f32_16x16x32_f16(
                    false, afrag, false, bf[nt], (short)0, acc[nt], false, false);
        }
    }

    // bias add (fp32) + store f16
    for (int nt = 0; nt < 12; nt++) {
        int ncol = nt * 16 + mrow;
        float bias = (nt < 4) ? bq[ncol] : (nt < 8) ? bk[ncol - 64] : bv[ncol - 128];
        for (int r = 0; r < 8; r++) {
            int row = tok0 + wv * 16 + r + ((lane < 16) ? 0 : 8);
            float val = acc[nt][r] + bias;
            if (nt < 4) {
                qb[(size_t)row * HEAD + ncol] = (_Float16)val;
            } else if (nt < 8) {
                kb[(size_t)row * HEAD + (ncol - 64)] = (_Float16)val;
            } else {
                int b = row >> 12, s = row & (SEQ - 1);
                vt[((size_t)b * HEAD + (ncol - 128)) * SEQ + s] = (_Float16)val;
            }
        }
    }
}

// ---------------------------------------------------------------------------
// Kernel 2: flash attention, causal, online softmax. One WG per 128-row
// q-block per batch; each wave owns 16 q rows. Double-buffered async
// global->LDS prefetch of K / V^T tiles (ASYNCcnt pipeline).
// ---------------------------------------------------------------------------
__global__ __launch_bounds__(256) void flash_attn(
    const _Float16* __restrict__ qb, const _Float16* __restrict__ kb,
    const _Float16* __restrict__ vt, float* __restrict__ out)
{
    __shared__ _Float16 lds_k[2][128 * 64];     // 2 x 16 KB: K tile [key][d]
    __shared__ _Float16 lds_v[2][64 * 128];     // 2 x 16 KB: V^T tile [d][key]
    __shared__ _Float16 lds_p[8][16 * 128];     // 32 KB: per-wave P [row][key]

    const int tid  = threadIdx.x;
    const int wv   = tid >> 5;
    const int lane = tid & 31;
    const int qblk = blockIdx.x;           // 0..31
    const int b    = blockIdx.y;           // 0..3
    const int qb0  = qblk * 128;
    const int mrow = lane & 15;
    const int hi   = (lane < 16) ? 0 : 8;  // row offset in C layout
    const int kb_off = (lane < 16) ? 0 : 16;

    // issue async prefetch of K / V^T tiles for key-block jb into buffer buf
    auto stage_tiles = [&](int jb, int buf) {
        const _Float16* ksrc = &kb[(size_t)(b * SEQ + jb * 128) * HEAD];
        for (int c = tid; c < 128 * 64 / 8; c += 256)              // 1024 x 16B
            async_copy16(ksrc + c * 8, &lds_k[buf][c * 8]);
        for (int c = tid; c < 64 * 128 / 8; c += 256) {            // 1024 x 16B
            int d = c >> 4, part = c & 15;
            async_copy16(&vt[((size_t)b * HEAD + d) * SEQ + jb * 128 + part * 8],
                         &lds_v[buf][d * 128 + part * 8]);
        }
    };

    stage_tiles(0, 0);   // prologue prefetch

    // Q fragments for this wave's 16 rows (overlaps with async prefetch)
    v16h qfrag[2];
    {
        size_t qrow = (size_t)(b * SEQ + qb0 + wv * 16 + mrow) * HEAD;
        for (int ks = 0; ks < 2; ks++)
            qfrag[ks] = load_afrag(&qb[qrow + ks * 32 + kb_off]);
    }

    v8f zero = {0.f, 0.f, 0.f, 0.f, 0.f, 0.f, 0.f, 0.f};
    v8f oacc[4];
    for (int i = 0; i < 4; i++) oacc[i] = zero;
    float rmax[8], rsum[8];
    for (int r = 0; r < 8; r++) { rmax[r] = -__builtin_inff(); rsum[r] = 0.f; }

    const float scale = 0.125f;  // 1/sqrt(64)

    for (int jb = 0; jb <= qblk; jb++) {
        const int cur = jb & 1;
        async_join();        // this wave's batch for block jb has landed
        __syncthreads();     // all waves' batches landed; prev compute done
        if (jb < qblk)
            stage_tiles(jb + 1, cur ^ 1);   // prefetch next block during compute

        // S = Q K^T  (8 key n-tiles x 2 k-steps)
        v8f sacc[8];
        for (int nt = 0; nt < 8; nt++) sacc[nt] = zero;
        for (int ks = 0; ks < 2; ks++) {
            v16h bf[8];
            for (int nt = 0; nt < 8; nt++)
                bf[nt] = *(const v16h*)&lds_k[cur][(nt * 16 + mrow) * 64 + ks * 32 + kb_off];
            for (int nt = 0; nt < 8; nt++)
                sacc[nt] = __builtin_amdgcn_wmma_f32_16x16x32_f16(
                    false, qfrag[ks], false, bf[nt], (short)0, sacc[nt], false, false);
        }

        // scale + causal mask (diagonal block only)
        const bool diag = (jb == qblk);
        for (int nt = 0; nt < 8; nt++) {
            int keyg = jb * 128 + nt * 16 + mrow;
            for (int r = 0; r < 8; r++) {
                float s = sacc[nt][r] * scale;
                if (diag) {
                    int mg = qb0 + wv * 16 + r + hi;
                    if (keyg > mg) s = -__builtin_inff();
                }
                sacc[nt][r] = s;
            }
        }

        // online softmax per row; P written to per-wave LDS in f16
        for (int r = 0; r < 8; r++) {
            float lm = sacc[0][r];
            for (int nt = 1; nt < 8; nt++) lm = fmaxf(lm, sacc[nt][r]);
            for (int m = 1; m < 16; m <<= 1) lm = fmaxf(lm, __shfl_xor(lm, m, 16));
            float mnew  = fmaxf(rmax[r], lm);
            float alpha = __expf(rmax[r] - mnew);
            rmax[r] = mnew;
            float psum = 0.f;
            int row = r + hi;
            for (int nt = 0; nt < 8; nt++) {
                float p = __expf(sacc[nt][r] - mnew);
                psum += p;
                lds_p[wv][row * 128 + nt * 16 + mrow] = (_Float16)p;
            }
            for (int m = 1; m < 16; m <<= 1) psum += __shfl_xor(psum, m, 16);
            rsum[r] = rsum[r] * alpha + psum;
            for (int d = 0; d < 4; d++) oacc[d][r] *= alpha;
        }

        // O += P x V   (4 k-steps over 128 keys x 4 d n-tiles)
        for (int ks = 0; ks < 4; ks++) {
            v16h afrag = load_afrag(&lds_p[wv][mrow * 128 + ks * 32 + kb_off]);
            v16h bf[4];
            for (int dt = 0; dt < 4; dt++)
                bf[dt] = *(const v16h*)&lds_v[cur][(dt * 16 + mrow) * 128 + ks * 32 + kb_off];
            for (int dt = 0; dt < 4; dt++)
                oacc[dt] = __builtin_amdgcn_wmma_f32_16x16x32_f16(
                    false, afrag, false, bf[dt], (short)0, oacc[dt], false, false);
        }
    }

    // normalize and store fp32 output [b][row][64]
    for (int r = 0; r < 8; r++) {
        float inv = 1.0f / rsum[r];
        int row = qb0 + wv * 16 + r + hi;
        size_t base = ((size_t)b * SEQ + row) * HEAD;
        for (int dt = 0; dt < 4; dt++) {
            out[base + dt * 16 + mrow] = oacc[dt][r] * inv;
        }
    }
}

// ---------------------------------------------------------------------------
extern "C" void kernel_launch(void* const* d_in, const int* in_sizes, int n_in,
                              void* d_out, int out_size, void* d_ws, size_t ws_size,
                              hipStream_t stream) {
    const float* x  = (const float*)d_in[0];
    const float* Wq = (const float*)d_in[1];
    const float* bq = (const float*)d_in[2];
    const float* Wk = (const float*)d_in[3];
    const float* bk = (const float*)d_in[4];
    const float* Wv = (const float*)d_in[5];
    const float* bv = (const float*)d_in[6];

    // workspace: Q, K as f16 [16384][64]; V^T as f16 [4][64][4096]  (6 MB)
    _Float16* qbuf = (_Float16*)d_ws;
    _Float16* kbuf = qbuf + (size_t)NB * SEQ * HEAD;
    _Float16* vbuf = kbuf + (size_t)NB * SEQ * HEAD;

    qkv_project<<<dim3(NB * SEQ / 128), dim3(256), 0, stream>>>(
        x, Wq, bq, Wk, bk, Wv, bv, qbuf, kbuf, vbuf);

    flash_attn<<<dim3(SEQ / 128, NB), dim3(256), 0, stream>>>(
        qbuf, kbuf, vbuf, (float*)d_out);
}
